// RGCN_75857712381962
// MI455X (gfx1250) — matrix operand order repaired and checked by reading, not compiled, not run on hardware
//
#include <hip/hip_runtime.h>
#include <hip/hip_bf16.h>
#include <stdint.h>

// ---------- problem constants (match reference) ----------
#define N0_C 262144
#define N1_C 16384
#define N2_C 1024
#define E0_C 409600
#define E1_C 20480
#define H_C  128
#define OUT_C 64
#define B_C  8
#define KTOT 1024   // B*H

// ---------- vector types (plain clang ext vectors: trivially constructible) ----------
typedef __attribute__((ext_vector_type(16))) __bf16        bf16x16;
typedef __attribute__((ext_vector_type(8)))  float         f32x8;
typedef __attribute__((ext_vector_type(4)))  float         f32x4;
typedef __attribute__((ext_vector_type(4)))  unsigned int  u32x4;
typedef __attribute__((ext_vector_type(4)))  unsigned short u16x4;

union FragAB { bf16x16 v; u32x4 q[2]; };

__device__ __forceinline__ unsigned short f2bf(float x) {
    unsigned int u = __float_as_uint(x);
    unsigned int r = (u + 0x7FFFu + ((u >> 16) & 1u)) >> 16;   // RNE
    return (unsigned short)r;
}

// hardware f32 atomic add, no return (global_atomic_add_f32, STOREcnt path)
__device__ __forceinline__ void atomic_add_f32(float* p, float v) {
    (void)__hip_atomic_fetch_add(p, v, __ATOMIC_RELAXED, __HIP_MEMORY_SCOPE_AGENT);
}

// ---------- zero workspace (float4 grid-stride) ----------
__global__ void __launch_bounds__(256) zero_f4(f32x4* __restrict__ p, long n4) {
    long i = (long)blockIdx.x * blockDim.x + threadIdx.x;
    long stride = (long)gridDim.x * blockDim.x;
    f32x4 z = {0.f, 0.f, 0.f, 0.f};
    for (; i < n4; i += stride) p[i] = z;
}

// ---------- transpose + f32->bf16: in [K x N] row-major -> out [N x K] ----------
__global__ void __launch_bounds__(256) transpose_cvt(const float* __restrict__ in,
                                                     unsigned short* __restrict__ out,
                                                     int K, int N) {
    int i = blockIdx.x * blockDim.x + threadIdx.x;
    if (i >= K * N) return;
    int kk = i / N;
    int n  = i - kk * N;
    out[(size_t)n * K + kk] = f2bf(in[i]);
}

// ---------- f32 -> bf16 bulk convert (4-wide, n divisible by 4) ----------
__global__ void __launch_bounds__(256) cvt_bf16x4(const f32x4* __restrict__ in,
                                                  u16x4* __restrict__ out, int n4) {
    int i = blockIdx.x * blockDim.x + threadIdx.x;
    int stride = gridDim.x * blockDim.x;
    for (; i < n4; i += stride) {
        f32x4 v = in[i];
        u16x4 o;
        o.x = f2bf(v.x); o.y = f2bf(v.y); o.z = f2bf(v.z); o.w = f2bf(v.w);
        out[i] = o;
    }
}

// ---------- edge scatter: one wave (32 lanes) per edge ----------
// agg[dst][b][0..127] += x[row][0..127] * comp[r][b] * norm    (f32 atomics -> L2)
__global__ void __launch_bounds__(256) edge_scatter(
    const int*   __restrict__ esrc, const int* __restrict__ edst,
    const int*   __restrict__ etype, const float* __restrict__ norm,
    const int*   __restrict__ src_ids,            // may be nullptr (direct index)
    const float* __restrict__ x,                  // [*, 128] rows
    const float* __restrict__ comp,               // [R, 8]
    float*       __restrict__ agg,                // [n_dst, 8, 128]
    int E) {
    int wid = (int)(((unsigned)blockIdx.x * blockDim.x + threadIdx.x) >> 5);
    if (wid >= E) return;
    int lane = threadIdx.x & 31;

    int   s   = esrc[wid];
    int   d   = edst[wid];
    int   r   = etype[wid];
    float nrm = norm[wid];
    long  row = src_ids ? (long)src_ids[s] : (long)s;

    f32x4 xv = *(const f32x4*)(x + row * 128 + lane * 4);
    float* base = agg + (size_t)d * (8 * 128) + lane * 4;
#pragma unroll
    for (int b = 0; b < 8; ++b) {
        float cb = comp[r * 8 + b] * nrm;
        float* p = base + b * 128;
        atomic_add_f32(p + 0, xv.x * cb);
        atomic_add_f32(p + 1, xv.y * cb);
        atomic_add_f32(p + 2, xv.z * cb);
        atomic_add_f32(p + 3, xv.w * cb);
    }
}

// ---------- WMMA GEMM: C[M,N] = A[M,K](bf16) * Bt[N,K](bf16)^T + bias, optional ReLU ----------
// One wave per 16x16 output tile; K stepped by 32 via v_wmma_f32_16x16x32_bf16.
// A fragment (16-bit 16x32): lane<16 -> M=lane, K {k0..k0+7, k0+16..k0+23};
//                            lane>=16 -> M=lane-16, K {k0+8..15, k0+24..31}.
// B fragment (16-bit 32x16): lane<16 -> N=lane, K {k0..k0+7, k0+16..k0+23}; lane>=16 -> N=lane-16, K {k0+8..15, k0+24..31}.
// C/D (f32 16x16): lanes<16 N=lane, vgpr p = row p; lanes>=16 N=lane-16, vgpr p = row 8+p.
__global__ void __launch_bounds__(256) wmma_gemm_bf16(
    const unsigned short* __restrict__ A,   // [M, K] bf16 bits, row-major
    const unsigned short* __restrict__ Bt,  // [N, K] bf16 bits, row-major (pre-transposed W)
    const float* __restrict__ bias,         // [N]
    float* __restrict__ C,                  // [M, N] f32 row-major
    int N, int K, int jtiles, int relu) {
    const int gw   = (int)(((unsigned)blockIdx.x * blockDim.x + threadIdx.x) >> 5);
    const int lane = threadIdx.x & 31;
    const int jt   = gw % jtiles;
    const int it   = gw / jtiles;
    const int half = lane >> 4;
    const int l    = lane & 15;

    const unsigned short* arow = A  + (size_t)(it * 16 + l) * K + half * 8;
    const unsigned short* brow = Bt + (size_t)(jt * 16 + l) * K + half * 8;

    f32x8 acc = {};
    for (int k0 = 0; k0 < K; k0 += 32) {
        __builtin_prefetch(arow + k0 + 256, 0, 3);   // near-scope global_prefetch_b8 on A stream
        FragAB a, b;
        a.q[0] = *(const u32x4*)(arow + k0);
        a.q[1] = *(const u32x4*)(arow + k0 + 16);
        b.q[0] = *(const u32x4*)(brow + k0);
        b.q[1] = *(const u32x4*)(brow + k0 + 16);
        acc = __builtin_amdgcn_wmma_f32_16x16x32_bf16(
            /*neg_a=*/false, a.v, /*neg_b=*/false, b.v,
            /*c_mod=*/(short)0, acc, /*reuse_a=*/false, /*reuse_b=*/false);
    }

    const int   n  = jt * 16 + l;
    const float bv = bias[n];
    const int   m0 = it * 16 + half * 8;
#pragma unroll
    for (int p = 0; p < 8; ++p) {
        float v = acc[p] + bv;
        if (relu) v = fmaxf(v, 0.0f);
        C[(size_t)(m0 + p) * N + n] = v;
    }
}

// ---------------------------------------------------------------------------
extern "C" void kernel_launch(void* const* d_in, const int* in_sizes, int n_in,
                              void* d_out, int out_size, void* d_ws, size_t ws_size,
                              hipStream_t stream) {
    const int*   src_ids = (const int*)  d_in[0];
    const int*   e0_src  = (const int*)  d_in[1];
    const int*   e0_dst  = (const int*)  d_in[2];
    const int*   e0_type = (const int*)  d_in[3];
    const float* norm0   = (const float*)d_in[4];
    const int*   e1_src  = (const int*)  d_in[5];
    const int*   e1_dst  = (const int*)  d_in[6];
    const int*   e1_type = (const int*)  d_in[7];
    const float* norm1   = (const float*)d_in[8];
    const float* emb     = (const float*)d_in[9];
    const float* V1      = (const float*)d_in[10];  // [8,128,128] == [1024,128] flat
    const float* comp1   = (const float*)d_in[11];  // [64,8]
    const float* b1      = (const float*)d_in[12];  // [128]
    const float* V2      = (const float*)d_in[13];  // [8,128,64] == [1024,64] flat
    const float* comp2   = (const float*)d_in[14];  // [64,8]
    const float* b2      = (const float*)d_in[15];  // [64]
    float*       out     = (float*)d_out;           // [1024,64]

    // ---- workspace layout (all segments 16B-aligned multiples) ----
    char* ws = (char*)d_ws;
    float*          agg1  = (float*)ws;                               // N1*1024 f32 = 64 MB
    float*          agg2  = agg1 + (size_t)N1_C * KTOT;               // N2*1024 f32 =  4 MB
    unsigned short* agg1b = (unsigned short*)(agg2 + (size_t)N2_C * KTOT); // 32 MB
    unsigned short* agg2b = agg1b + (size_t)N1_C * KTOT;              //  2 MB
    float*          h     = (float*)(agg2b + (size_t)N2_C * KTOT);    // N1*128 f32 = 8 MB
    unsigned short* wt1   = (unsigned short*)(h + (size_t)N1_C * H_C);// 128*1024 bf16
    unsigned short* wt2   = wt1 + (size_t)H_C * KTOT;                 // 64*1024 bf16

    // 1) zero both accumulators (contiguous region)
    {
        long n4 = ((long)N1_C * KTOT + (long)N2_C * KTOT) / 4;
        zero_f4<<<2048, 256, 0, stream>>>((f32x4*)agg1, n4);
    }
    // 2) transpose+convert weights: Wt[n][kk] = V[kk][n] in bf16
    transpose_cvt<<<(KTOT * H_C + 255) / 256, 256, 0, stream>>>(V1, wt1, KTOT, H_C);
    transpose_cvt<<<(KTOT * OUT_C + 255) / 256, 256, 0, stream>>>(V2, wt2, KTOT, OUT_C);

    // 3) layer-1 edge scatter: gather emb[src_ids[e0_src]] -> atomic agg1[dst,b,:]
    edge_scatter<<<E0_C / 8, 256, 0, stream>>>(e0_src, e0_dst, e0_type, norm0,
                                               src_ids, emb, comp1, agg1, E0_C);

    // 4) agg1 -> bf16
    cvt_bf16x4<<<4096, 256, 0, stream>>>((const f32x4*)agg1, (u16x4*)agg1b,
                                         (int)((size_t)N1_C * KTOT / 4));

    // 5) GEMM1: h = relu(agg1b[16384,1024] @ W1[1024,128] + b1)   (8192 tiles)
    wmma_gemm_bf16<<<(N1_C / 16) * (H_C / 16) / 8, 256, 0, stream>>>(
        agg1b, wt1, b1, h, H_C, KTOT, H_C / 16, /*relu=*/1);

    // 6) layer-2 edge scatter: gather h[e1_src] -> atomic agg2[dst,b,:]
    edge_scatter<<<E1_C / 8, 256, 0, stream>>>(e1_src, e1_dst, e1_type, norm1,
                                               nullptr, h, comp2, agg2, E1_C);

    // 7) agg2 -> bf16
    cvt_bf16x4<<<1024, 256, 0, stream>>>((const f32x4*)agg2, (u16x4*)agg2b,
                                         (int)((size_t)N2_C * KTOT / 4));

    // 8) GEMM2: out = agg2b[1024,1024] @ W2[1024,64] + b2   (256 tiles)
    wmma_gemm_bf16<<<(N2_C / 16) * (OUT_C / 16) / 8, 256, 0, stream>>>(
        agg2b, wt2, b2, out, OUT_C, KTOT, OUT_C / 16, /*relu=*/0);
}
